// MultiHeadLinearAttention_28707561407083
// MI455X (gfx1250) — matrix operand order repaired and checked by reading, not compiled, not run on hardware
//
#include <hip/hip_runtime.h>
#include <hip/hip_bf16.h>

// MultiHeadLinearAttention for MI455X (gfx1250), bf16 WMMA pipeline.
// B=4, L=8192, F=1024, H=16, D=64.

typedef __bf16 bf16;
typedef __attribute__((ext_vector_type(16))) __bf16 v16bf;
typedef __attribute__((ext_vector_type(8)))  __bf16 v8bf;
typedef __attribute__((ext_vector_type(8)))  float  v8f;
typedef __attribute__((ext_vector_type(4)))  float  v4f;
typedef __attribute__((ext_vector_type(4)))  int    v4i;

union Frag {            // 16 bf16 = one WMMA A/B operand (8 VGPRs)
    v16bf v;
    v8bf  h[2];
    bf16  e[16];
};

#define BATCH 4
#define SEQL  8192
#define FDIM  1024
#define NHEAD 16
#define HDIM  64
#define BL    (BATCH * SEQL)        // 32768
#define HD    (NHEAD * HDIM)        // 1024
#define KV_SPLIT 16
#define KV_NPAD  80                 // 64 e-cols + 16 "ones" cols (ksum fold-in)

// CDNA5 async global->LDS staging (guarded; falls back to VGPR path).
// Probe-learned signature: (v4i as1*, v4i as3*, imm offset, imm cpol);
// the offset applies to BOTH the global and LDS addresses (ISA 15.18.3).
#if defined(__gfx1250__) && __has_builtin(__builtin_amdgcn_global_load_async_to_lds_b128)
#define ASYNC_LDS 1
typedef __attribute__((address_space(1))) v4i* gas_ptr;
typedef __attribute__((address_space(3))) v4i* las_ptr;
#else
#define ASYNC_LDS 0
#endif

__device__ __forceinline__ void wait_async0() {
#if defined(__gfx1250__)
#if __has_builtin(__builtin_amdgcn_s_wait_asynccnt)
    __builtin_amdgcn_s_wait_asynccnt(0);
#else
    asm volatile("s_wait_asynccnt 0x0" ::: "memory");
#endif
#endif
}

// ---------------------------------------------------------------------------
// K0: convert four 1Mi-element f32 [K][N] weights to bf16 TRANSPOSED [N][K].
// One-time 8MB output; scattered writes absorbed by L2.
// ---------------------------------------------------------------------------
__global__ __launch_bounds__(256) void wconv_t(const float* __restrict__ a,
                                               const float* __restrict__ b,
                                               const float* __restrict__ c,
                                               const float* __restrict__ d,
                                               bf16* __restrict__ out) {
    unsigned idx = blockIdx.x * 256u + threadIdx.x;      // 0 .. 4Mi-1
    unsigned sel = idx >> 20, off = idx & 0xFFFFFu;
    unsigned k = off >> 10, n = off & 1023u;
    float v = (sel == 0) ? a[off] : (sel == 1) ? b[off] : (sel == 2) ? c[off] : d[off];
    out[(sel << 20) | (n << 10) | k] = (bf16)v;
}

__device__ __forceinline__ float phi_fn(float x) {       // elu(x)+1
    return x > 0.0f ? x + 1.0f : __expf(x);
}

// ---------------------------------------------------------------------------
// K1/K4: tiled GEMM  C[M,N] = op(A[M,K] @ W[K,N]), W given transposed [N][K].
// 256 threads (8 waves), tile 128x128, k-step 32, double-buffered LDS,
// one barrier per k-step. A and B staging share the same thread map
// (contiguous 32B per thread). Async global->LDS used where data is bf16.
// Each wave owns a 64x32 sub-tile = 4x2 accumulators of 16x16.
// ---------------------------------------------------------------------------
template <typename AT, typename OT, bool PHI>
__global__ __launch_bounds__(256)
void gemm_proj(const AT* __restrict__ A, const bf16* __restrict__ Wt,
               OT* __restrict__ C, int M, int N, int K) {
    __shared__ bf16 As[2][128][40];    // [buf][m][k], padded stride
    __shared__ bf16 Bs[2][128][40];    // [buf][n][k], padded stride

    const int tid  = threadIdx.x;
    const int lane = tid & 31;
    const int wave = tid >> 5;
    const int wm   = (wave & 1) * 64;
    const int wn   = (wave >> 1) * 32;
    const int mBase = blockIdx.x * 128;
    const int nBase = blockIdx.y * 128;
    const int hi = lane >> 4;              // half-wave select
    const int lr = lane & 15;

    const int row  = tid >> 1;             // staging map: 128 rows x 2 halves
    const int half = (tid & 1) * 16;

    const AT*   aPtr = A  + (size_t)(mBase + row) * K + half;
    const bf16* bPtr = Wt + (size_t)(nBase + row) * K + half;

    constexpr bool A_BF = (sizeof(AT) == 2);
    constexpr bool A_ASYNC = A_BF && (ASYNC_LDS != 0);
    constexpr bool B_ASYNC = (ASYNC_LDS != 0);

    union Pack { v8bf h[2]; bf16 e[16]; };
    Pack ta, tb;

    auto loadA = [&](int k0) {
        if constexpr (!A_BF) {
            v4f x0 = *(const v4f*)(aPtr + k0);
            v4f x1 = *(const v4f*)(aPtr + k0 + 4);
            v4f x2 = *(const v4f*)(aPtr + k0 + 8);
            v4f x3 = *(const v4f*)(aPtr + k0 + 12);
            #pragma unroll
            for (int j = 0; j < 4; ++j) {
                ta.e[j]      = (bf16)x0[j];
                ta.e[4 + j]  = (bf16)x1[j];
                ta.e[8 + j]  = (bf16)x2[j];
                ta.e[12 + j] = (bf16)x3[j];
            }
        } else {
            ta.h[0] = *(const v8bf*)(aPtr + k0);
            ta.h[1] = *(const v8bf*)(aPtr + k0 + 8);
        }
    };
    auto loadB = [&](int k0) {
        tb.h[0] = *(const v8bf*)(bPtr + k0);
        tb.h[1] = *(const v8bf*)(bPtr + k0 + 8);
    };
    auto storeA = [&](int buf) {
        *(v8bf*)&As[buf][row][half]     = ta.h[0];
        *(v8bf*)&As[buf][row][half + 8] = ta.h[1];
    };
    auto storeB = [&](int buf) {
        *(v8bf*)&Bs[buf][row][half]     = tb.h[0];
        *(v8bf*)&Bs[buf][row][half + 8] = tb.h[1];
    };
#if ASYNC_LDS
    auto asyncA = [&](int buf, int k0) {
        gas_ptr g = (gas_ptr)(v4i*)(uintptr_t)(aPtr + k0);
        las_ptr l = (las_ptr)(v4i*)&As[buf][row][half];
        __builtin_amdgcn_global_load_async_to_lds_b128(g, l, 0, 0);
        __builtin_amdgcn_global_load_async_to_lds_b128(g, l, 16, 0);
    };
    auto asyncB = [&](int buf, int k0) {
        gas_ptr g = (gas_ptr)(v4i*)(uintptr_t)(bPtr + k0);
        las_ptr l = (las_ptr)(v4i*)&Bs[buf][row][half];
        __builtin_amdgcn_global_load_async_to_lds_b128(g, l, 0, 0);
        __builtin_amdgcn_global_load_async_to_lds_b128(g, l, 16, 0);
    };
#endif

    v8f acc[4][2] = {};

    // prologue: stage tile 0 into buffer 0
#if ASYNC_LDS
    if constexpr (A_ASYNC) asyncA(0, 0); else { loadA(0); storeA(0); }
    asyncB(0, 0);
    wait_async0();
#else
    loadA(0); storeA(0);
    loadB(0); storeB(0);
#endif
    __syncthreads();

    const int nk = K / 32;
    for (int t = 0; t < nk; ++t) {
        const int cur = t & 1, nxt = cur ^ 1;
        const bool pre = (t + 1 < nk);
        // issue next tile's loads before the math
        if (pre) {
#if ASYNC_LDS
            if constexpr (A_ASYNC) asyncA(nxt, (t + 1) * 32); else loadA((t + 1) * 32);
            asyncB(nxt, (t + 1) * 32);
#else
            loadA((t + 1) * 32);
            loadB((t + 1) * 32);
#endif
        }

        // fragments (ISA 7.12.2 bf16 layouts) + 8 WMMAs from buffer `cur`
        Frag af[4], bfm[2];
        const int akoff = hi ? 8 : 0;     // A lanes 16-31 hold K=8..15,24..31
        #pragma unroll
        for (int mi = 0; mi < 4; ++mi) {
            const bf16* p = &As[cur][wm + mi * 16 + lr][akoff];
            af[mi].h[0] = *(const v8bf*)(p);
            af[mi].h[1] = *(const v8bf*)(p + 16);
        }
        const int bkoff = hi ? 8 : 0;     // B lanes 16-31 hold K=8..15,24..31
        #pragma unroll
        for (int ni = 0; ni < 2; ++ni) {
            const bf16* p = &Bs[cur][wn + ni * 16 + lr][bkoff];
            bfm[ni].h[0] = *(const v8bf*)(p);
            bfm[ni].h[1] = *(const v8bf*)(p + 16);
        }
        #pragma unroll
        for (int mi = 0; mi < 4; ++mi)
            #pragma unroll
            for (int ni = 0; ni < 2; ++ni)
                acc[mi][ni] = __builtin_amdgcn_wmma_f32_16x16x32_bf16(
                    false, af[mi].v, false, bfm[ni].v, (short)0, acc[mi][ni],
                    false, false);

        if (pre) {
            if constexpr (!A_ASYNC) storeA(nxt);
            if constexpr (!B_ASYNC) storeB(nxt);
#if ASYNC_LDS
            wait_async0();
#endif
        }
        __syncthreads();
    }

    // epilogue: optional phi, convert, store
    #pragma unroll
    for (int mi = 0; mi < 4; ++mi)
        #pragma unroll
        for (int ni = 0; ni < 2; ++ni)
            #pragma unroll
            for (int r = 0; r < 8; ++r) {
                int m = mBase + wm + mi * 16 + r + hi * 8;
                int n = nBase + wn + ni * 16 + lr;
                float v = acc[mi][ni][r];
                if constexpr (PHI) v = phi_fn(v);
                C[(size_t)m * N + n] = (OT)v;
            }
}

// ---------------------------------------------------------------------------
// K2: kv[b,h,d,e] = sum_l phi(k)[l,d] * v[l,e]   (plus ksum via ones-columns)
// One wave per (b*h, split); A = kp^T (4 m-tiles of d), B = [v | ones | 0]
// (5 n-tiles, col 64 = 1). Partials reduced deterministically in K2b.
// ---------------------------------------------------------------------------
__global__ __launch_bounds__(32)
void kv_state(const bf16* __restrict__ kp, const bf16* __restrict__ vv,
              float* __restrict__ part) {
    __shared__ bf16 kpT[64][40];           // [d][l]
    __shared__ bf16 vT[KV_NPAD][40];       // [e][l], rows 64..79 constant
    const int bh = blockIdx.x;             // 0..63
    const int b = bh >> 4, h = bh & 15;
    const int split = blockIdx.y;          // 0..KV_SPLIT-1
    const int lane = threadIdx.x;
    const int hi = lane >> 4, lr = lane & 15;

    #pragma unroll
    for (int j = 0; j < 16; ++j)
        vT[64 + j][lane] = (bf16)((j == 0) ? 1.0f : 0.0f);

    v8f acc[4][5] = {};

    for (int it = 0; it < 512 / 32; ++it) {
        int l = split * 512 + it * 32 + lane;
        size_t base = ((size_t)(b * SEQL + l)) * HD + h * HDIM;
        union { v8bf h8[8]; bf16 e[64]; } kr, vr;
        #pragma unroll
        for (int j = 0; j < 8; ++j) kr.h8[j] = *(const v8bf*)(kp + base + j * 8);
        #pragma unroll
        for (int j = 0; j < 8; ++j) vr.h8[j] = *(const v8bf*)(vv + base + j * 8);

        __syncthreads();                    // prior frag reads done
        #pragma unroll
        for (int d = 0; d < 64; ++d) kpT[d][lane] = kr.e[d];
        #pragma unroll
        for (int e = 0; e < 64; ++e) vT[e][lane] = vr.e[e];
        __syncthreads();

        Frag af[4], bfm[5];
        #pragma unroll
        for (int mi = 0; mi < 4; ++mi) {
            const bf16* p = &kpT[mi * 16 + lr][hi * 8];
            af[mi].h[0] = *(const v8bf*)(p);
            af[mi].h[1] = *(const v8bf*)(p + 16);
        }
        #pragma unroll
        for (int ni = 0; ni < 5; ++ni) {
            const bf16* p = &vT[ni * 16 + lr][hi * 8];
            bfm[ni].h[0] = *(const v8bf*)(p);
            bfm[ni].h[1] = *(const v8bf*)(p + 16);
        }
        #pragma unroll
        for (int mi = 0; mi < 4; ++mi)
            #pragma unroll
            for (int ni = 0; ni < 5; ++ni)
                acc[mi][ni] = __builtin_amdgcn_wmma_f32_16x16x32_bf16(
                    false, af[mi].v, false, bfm[ni].v, (short)0, acc[mi][ni],
                    false, false);
    }

    float* out = part + ((size_t)(bh * KV_SPLIT + split)) * (64 * KV_NPAD);
    #pragma unroll
    for (int mi = 0; mi < 4; ++mi)
        #pragma unroll
        for (int ni = 0; ni < 5; ++ni)
            #pragma unroll
            for (int r = 0; r < 8; ++r) {
                int d = mi * 16 + r + hi * 8;
                int c = ni * 16 + lr;
                out[d * KV_NPAD + c] = acc[mi][ni][r];
            }
}

// K2b: reduce partials; emit kv TRANSPOSED as bf16 [e][d] (B-frag friendly)
// and ksum as f32.
__global__ __launch_bounds__(256)
void kv_reduce(const float* __restrict__ part, bf16* __restrict__ kvt,
               float* __restrict__ ksum) {
    int bh = blockIdx.x;
    for (int i = threadIdx.x; i < 64 * 65; i += 256) {
        int d = i / 65, c = i % 65;
        float s = 0.0f;
        for (int sp = 0; sp < KV_SPLIT; ++sp)
            s += part[((size_t)(bh * KV_SPLIT + sp)) * (64 * KV_NPAD) + d * KV_NPAD + c];
        if (c < 64) kvt[(size_t)bh * 4096 + c * 64 + d] = (bf16)s;   // [e][d]
        else        ksum[bh * 64 + d] = s;
    }
}

// ---------------------------------------------------------------------------
// K3: out_t[l, h*64+e] = z_l * sum_d qp[l,d] * kv[d,e];  z = 1/(qp.ksum+eps).
// One wave per (16-row tile, b*h). K=64 -> 2 WMMA steps, N=64 -> 4 tiles.
// kv comes in transposed bf16 [e][d] so B fragments are contiguous loads.
// ---------------------------------------------------------------------------
__global__ __launch_bounds__(32)
void attn_apply(const bf16* __restrict__ qp, const bf16* __restrict__ kvt,
                const float* __restrict__ ksum, bf16* __restrict__ outt) {
    const int bh = blockIdx.y, b = bh >> 4, h = bh & 15;
    const int l0 = blockIdx.x * 16;
    const int lane = threadIdx.x;
    const int hi = lane >> 4, lr = lane & 15;
    const bf16* kv = kvt + (size_t)bh * 4096;
    const float* ks = ksum + bh * 64;

    // denominator: each lane does half of one row's 64-dot, then fold halves.
    size_t rowbase = ((size_t)(b * SEQL + l0 + lr)) * HD + h * HDIM;
    const bf16* qr = qp + rowbase + hi * 32;
    float den = 0.0f;
    #pragma unroll
    for (int c = 0; c < 4; ++c) {
        v8bf q = *(const v8bf*)(qr + c * 8);
        #pragma unroll
        for (int j = 0; j < 8; ++j)
            den += (float)q[j] * ks[hi * 32 + c * 8 + j];
    }
    den += __shfl_xor(den, 16, 32);
    float z = 1.0f / (den + 1e-6f);

    v8f acc[4] = {};
    #pragma unroll
    for (int s = 0; s < 2; ++s) {
        Frag af;
        size_t p = rowbase + s * 32 + hi * 8;
        af.h[0] = *(const v8bf*)(qp + p);
        af.h[1] = *(const v8bf*)(qp + p + 16);
        #pragma unroll
        for (int ni = 0; ni < 4; ++ni) {
            Frag bfm;
            const bf16* bp = kv + (ni * 16 + lr) * 64 + s * 32 + hi * 8;
            bfm.h[0] = *(const v8bf*)(bp);
            bfm.h[1] = *(const v8bf*)(bp + 16);
            acc[ni] = __builtin_amdgcn_wmma_f32_16x16x32_bf16(
                false, af.v, false, bfm.v, (short)0, acc[ni], false, false);
        }
    }
    #pragma unroll
    for (int ni = 0; ni < 4; ++ni)
        #pragma unroll
        for (int r = 0; r < 8; ++r) {
            int m = r + hi * 8;
            float zm = __shfl(z, m, 32);
            int row = b * SEQL + l0 + m;
            int col = h * HDIM + ni * 16 + lr;
            outt[(size_t)row * HD + col] = (bf16)(acc[ni][r] * zm);
        }
}

// ---------------------------------------------------------------------------
extern "C" void kernel_launch(void* const* d_in, const int* in_sizes, int n_in,
                              void* d_out, int out_size, void* d_ws, size_t ws_size,
                              hipStream_t stream) {
    const float* xq  = (const float*)d_in[0];   // [B,L,F]
    const float* xkv = (const float*)d_in[1];   // [B,L,F]
    const float* Wq  = (const float*)d_in[2];   // [F,H,D] = [1024,1024]
    const float* Wk  = (const float*)d_in[3];
    const float* Wv  = (const float*)d_in[4];
    const float* Wo  = (const float*)d_in[5];   // [H,D,F] = [1024,1024]

    // Workspace carve-up (~221 MB).
    const size_t WSZ = (size_t)1 << 20;          // 1Mi elements per weight
    const size_t ACT = (size_t)BL * HD;          // 32768*1024
    bf16*  wqb  = (bf16*)d_ws;                   // transposed [N][K]
    bf16*  wkb  = wqb + WSZ;
    bf16*  wvb  = wkb + WSZ;
    bf16*  wob  = wvb + WSZ;
    bf16*  qp   = wob + WSZ;
    bf16*  kp   = qp + ACT;
    bf16*  vb   = kp + ACT;
    float* part = (float*)(vb + ACT);            // 64*16*64*80 f32 = 20 MiB
    bf16*  kvt  = (bf16*)(part + (size_t)64 * KV_SPLIT * 64 * KV_NPAD);
    float* ksm  = (float*)(kvt + (size_t)64 * 4096);
    bf16*  outt = kp;                            // kp dead after kv_state

    // K0: weights -> bf16, transposed
    wconv_t<<<16384, 256, 0, stream>>>(Wq, Wk, Wv, Wo, wqb);

    // K1: fused projections + phi (q,k), plain (v)
    dim3 g(BL / 128, HD / 128);
    gemm_proj<float, bf16, true ><<<g, 256, 0, stream>>>(xq,  wqb, qp, BL, HD, FDIM);
    gemm_proj<float, bf16, true ><<<g, 256, 0, stream>>>(xkv, wkb, kp, BL, HD, FDIM);
    gemm_proj<float, bf16, false><<<g, 256, 0, stream>>>(xkv, wvb, vb, BL, HD, FDIM);

    // K2: kv state + ksum (ones-column trick), two-stage deterministic reduce
    kv_state<<<dim3(64, KV_SPLIT), 32, 0, stream>>>(kp, vb, part);
    kv_reduce<<<64, 256, 0, stream>>>(part, kvt, ksm);

    // K3: qp @ kv, scaled by 1/(qp.ksum + eps)
    attn_apply<<<dim3(SEQL / 16, 64), 32, 0, stream>>>(qp, kvt, ksm, outt);

    // K4: output projection -> f32 d_out
    gemm_proj<bf16, float, false><<<g, 256, 0, stream>>>(outt, wob, (float*)d_out,
                                                         BL, FDIM, HD);
}
